// WordAttention_4209067950408
// MI455X (gfx1250) — matrix operand order repaired and checked by reading, not compiled, not run on hardware
//
#include <hip/hip_runtime.h>

// Problem constants (B=8, S=2048, D=128, H=4, HD=32)
#define BB 8
#define SS 2048
#define DD 128
#define HH 4
#define HD 32

typedef __attribute__((ext_vector_type(16))) __bf16          v16bf;
typedef __attribute__((ext_vector_type(16))) float           v16f;
typedef __attribute__((ext_vector_type(8)))  float           v8f;
typedef __attribute__((ext_vector_type(4)))  unsigned int    v4u;
typedef __attribute__((ext_vector_type(4)))  float           v4f;
typedef __attribute__((ext_vector_type(4)))  unsigned int    u32x4;
typedef __attribute__((ext_vector_type(8)))  int             i32x8;
typedef __attribute__((ext_vector_type(4)))  int             i32x4;

union V16 {                 // one WMMA 16-bit operand (8 VGPRs / lane)
    v16bf b;
    v4u   q[2];
};
union VF16 {                // 16 floats for packed conversion
    v16f f;
    v4f  q[4];
};

__device__ __forceinline__ v8f wmma_bf16(v16bf a, v16bf b, v8f c) {
    return __builtin_amdgcn_wmma_f32_16x16x32_bf16(
        /*neg_a=*/false, a, /*neg_b=*/false, b,
        /*c_mod=*/(short)0, c, /*reuse_a=*/false, /*reuse_b=*/false);
}

#if __has_builtin(__builtin_amdgcn_tensor_load_to_lds) && \
    __has_builtin(__builtin_amdgcn_s_wait_tensorcnt)
#define HAVE_TDM 1
#else
#define HAVE_TDM 0
#endif

#if HAVE_TDM
// 2D bf16 tile DMA: global (tile1 rows x tile0 elems, row stride stride0 elems)
// -> LDS (packed row-major). D# layout per CDNA5 ISA ch.8 (group0 128b, group1 256b).
// This toolchain exposes the 6-arg builtin: (g0, g1, g2, g3, g4, cpol).
__device__ __forceinline__ void tdm_load_2d(unsigned lds_addr, const void* gptr,
                                            unsigned tile0, unsigned tile1,
                                            unsigned long long stride0)
{
    unsigned long long ga = (unsigned long long)(size_t)gptr;
    u32x4 g0;
    g0[0] = 1u;                                   // count=1 (valid), user mode
    g0[1] = lds_addr;                             // LDS byte address
    g0[2] = (unsigned)ga;                         // global_addr[31:0]
    g0[3] = ((unsigned)(ga >> 32) & 0x01FFFFFFu)  // global_addr[56:32]
            | (2u << 30);                         // type = 2 ("image")
    i32x8 g1;
    g1[0] = 1 << 16;                              // data_size = 1 -> 2 bytes
    g1[1] = (int)((tile0 & 0xFFFFu) << 16);       // tensor_dim0[15:0] (== tile0)
    g1[2] = (int)((tile0 >> 16) | ((tile1 & 0xFFFFu) << 16));   // dim0 hi | dim1 lo
    g1[3] = (int)((tile1 >> 16) | (tile0 << 16)); // dim1 hi | tile_dim0
    g1[4] = (int)tile1;                           // tile_dim1 | tile_dim2=0
    g1[5] = (int)(unsigned)stride0;               // tensor_dim0_stride[31:0]
    g1[6] = (int)((unsigned)(stride0 >> 32) & 0xFFFFu);  // stride0 hi | stride1 lo=0
    g1[7] = 0;
    i32x4 gz4 = {0, 0, 0, 0};
    i32x8 gz8 = {0, 0, 0, 0, 0, 0, 0, 0};
    __builtin_amdgcn_tensor_load_to_lds(g0, g1, gz4, gz4, gz8, 0);
}
#endif

// ---------------------------------------------------------------------------
// Kernel 1: qkv = x @ w_attn  (M=16384, K=128, N=384), one wave per 16x16 tile.
// Writes Q,K as bf16 [B,H,S,HD]; V transposed as bf16 [B,H,HD,S].
// ---------------------------------------------------------------------------
__global__ __launch_bounds__(256) void qkv_kernel(
    const float* __restrict__ x, const float* __restrict__ w_attn,
    __bf16* __restrict__ Qg, __bf16* __restrict__ Kg, __bf16* __restrict__ Vt)
{
    const int wv   = blockIdx.x * (blockDim.x >> 5) + (threadIdx.x >> 5);
    const int mt   = wv / 24;            // 1024 M-tiles
    const int nt   = wv % 24;            // 24  N-tiles
    const int m0   = mt * 16, n0 = nt * 16;
    const int lane = threadIdx.x & 31;
    const int half = lane >> 4;
    const int n    = lane & 15;

    v8f acc = {};
    const float* xr = x + (size_t)(m0 + (lane & 15)) * DD;
    const float* wc = w_attn + n0 + n;          // column base, stride 3*DD
#pragma unroll
    for (int kc = 0; kc < 4; ++kc) {
        VF16 af;
        af.q[0] = *(const v4f*)(xr + kc * 32 + half * 16);
        af.q[1] = *(const v4f*)(xr + kc * 32 + half * 16 + 4);
        af.q[2] = *(const v4f*)(xr + kc * 32 + half * 16 + 8);
        af.q[3] = *(const v4f*)(xr + kc * 32 + half * 16 + 12);
        V16 a;
        a.b = __builtin_convertvector(af.f, v16bf);
        VF16 bf;
#pragma unroll
        for (int e = 0; e < 16; ++e)
            bf.f[e] = wc[(size_t)(kc * 32 + half * 16 + e) * (3 * DD)];
        V16 bm;
        bm.b = __builtin_convertvector(bf.f, v16bf);
        acc = wmma_bf16(a.b, bm.b, acc);
    }

    // Q/K/V region and head index are tile-uniform (n0 multiple of 16, heads 32 wide).
    const int bb     = m0 >> 11;
    const int s_base = (m0 & (SS - 1)) + 8 * half;
    if (n0 < DD) {
        const int h  = n0 >> 5;
        const int hd = (n0 & 31) + n;
        __bf16* p = Qg + ((size_t)(bb * HH + h) * SS + s_base) * HD + hd;
#pragma unroll
        for (int r = 0; r < 8; ++r) p[r * HD] = (__bf16)acc[r];
    } else if (n0 < 2 * DD) {
        const int c0 = n0 - DD;
        const int h  = c0 >> 5;
        const int hd = (c0 & 31) + n;
        __bf16* p = Kg + ((size_t)(bb * HH + h) * SS + s_base) * HD + hd;
#pragma unroll
        for (int r = 0; r < 8; ++r) p[r * HD] = (__bf16)acc[r];
    } else {
        const int c0 = n0 - 2 * DD;
        const int h  = c0 >> 5;
        const int hd = (c0 & 31) + n;
        __bf16* p = Vt + ((size_t)(bb * HH + h) * HD + hd) * SS + s_base;
#pragma unroll
        for (int r = 0; r < 8; ++r) p[r] = (__bf16)acc[r];
    }
}

// ---------------------------------------------------------------------------
// Kernel 2: causal flash attention.
// One block = (b, h, 128 consecutive queries) = 8 waves x one 16-query tile.
// K/V 32-key tiles are staged into LDS once per block via the Tensor Data
// Mover (TENSOR_LOAD_TO_LDS + s_wait_tensorcnt), then consumed by all 8 waves.
// ---------------------------------------------------------------------------
__global__ __launch_bounds__(256) void attn_kernel(
    const __bf16* __restrict__ Qg, const __bf16* __restrict__ Kg,
    const __bf16* __restrict__ Vt, __bf16* __restrict__ Og)
{
    __shared__ __bf16 kT[32][32];          // [key][hd]   2 KB
    __shared__ __bf16 vT[32][32];          // [hd][key]   2 KB
    __shared__ __bf16 p_lds[8][16 * 32];   // per-wave P tile, 8 KB

    // block -> (b, h, q-block of 128)
    const int b     = blockIdx.x >> 6;            // / (H * S/128) = /64
    const int rem   = blockIdx.x & 63;
    const int h     = rem >> 4;
    const int q0blk = (rem & 15) * 128;
    const int widx  = threadIdx.x >> 5;
    const int q0    = q0blk + widx * 16;          // this wave's query tile
    const int lane  = threadIdx.x & 31;
    const int half  = lane >> 4;
    const int n     = lane & 15;
    const int bh    = b * HH + h;

    const __bf16* Qb = Qg + (size_t)bh * SS * HD;
    const __bf16* Kb = Kg + (size_t)bh * SS * HD;
    const __bf16* Vb = Vt + (size_t)bh * HD * SS;

    // Q tile in A-layout registers (row m = lane&15)
    V16 aq;
    {
        const __bf16* qr = Qb + (size_t)(q0 + (lane & 15)) * HD;
        aq.q[0] = *(const v4u*)(qr + half * 16);
        aq.q[1] = *(const v4u*)(qr + half * 16 + 8);
    }

    v8f acc0 = {}, acc1 = {};
    float mrun[8], lrun[8];
#pragma unroll
    for (int r = 0; r < 8; ++r) { mrun[r] = -3.0e30f; lrun[r] = 0.0f; }
    const float scale = 0.17677669529663688f;   // 1/sqrt(32)

    const int kmax = q0blk + 127;               // largest causal key for this block
    for (int k0 = 0; k0 <= kmax; k0 += 32) {
        // ---- stage K (32x32, rows=keys) and V (32x32, rows=hd) tiles into LDS
#if HAVE_TDM
        if (widx == 0) {
            tdm_load_2d((unsigned)(size_t)&kT[0][0], Kb + (size_t)k0 * HD,
                        /*tile0=*/HD, /*tile1=*/32, /*stride0=*/HD);
            tdm_load_2d((unsigned)(size_t)&vT[0][0], Vb + k0,
                        /*tile0=*/32, /*tile1=*/HD, /*stride0=*/SS);
            __builtin_amdgcn_s_wait_tensorcnt(0);
        }
#else
        {
            const int t = threadIdx.x;
            if (t < 128) {                       // K tile: 128 threads x 16B
                int row = t >> 2, col = (t & 3) * 8;
                *(v4u*)&kT[row][col] = *(const v4u*)(Kb + (size_t)(k0 + row) * HD + col);
            } else {                             // V tile
                int tt = t - 128; int row = tt >> 2, col = (tt & 3) * 8;
                *(v4u*)&vT[row][col] = *(const v4u*)(Vb + (size_t)row * SS + k0 + col);
            }
        }
#endif
        __syncthreads();

        // ---- wave-uniform causal skip: this wave needs keys only up to q0+15
        if (k0 <= q0 + 15) {
            V16 bk0, bk1;
            bk0.q[0] = ((const v4u*)&kT[n][half * 16])[0];
            bk0.q[1] = ((const v4u*)&kT[n][half * 16])[1];
            bk1.q[0] = ((const v4u*)&kT[16 + n][half * 16])[0];
            bk1.q[1] = ((const v4u*)&kT[16 + n][half * 16])[1];

            v8f zero = {};
            v8f s0 = wmma_bf16(aq.b, bk0.b, zero);
            v8f s1 = wmma_bf16(aq.b, bk1.b, zero);

#pragma unroll
            for (int r = 0; r < 8; ++r) {
                int q = q0 + r + 8 * half;
                float x0 = s0[r] * scale; if (k0 + n > q)       x0 = -3.0e30f;
                float x1 = s1[r] * scale; if (k0 + 16 + n > q)  x1 = -3.0e30f;
                s0[r] = x0; s1[r] = x1;
            }

            float tm[8];
#pragma unroll
            for (int r = 0; r < 8; ++r) tm[r] = fmaxf(s0[r], s1[r]);
#pragma unroll
            for (int off = 1; off < 16; off <<= 1) {
#pragma unroll
                for (int r = 0; r < 8; ++r)
                    tm[r] = fmaxf(tm[r], __shfl_xor(tm[r], off, 32));
            }

            float corr[8], rs[8];
#pragma unroll
            for (int r = 0; r < 8; ++r) {
                float nm = fmaxf(mrun[r], tm[r]);
                corr[r]  = __expf(mrun[r] - nm);
                mrun[r]  = nm;
                s0[r] = __expf(s0[r] - nm);
                s1[r] = __expf(s1[r] - nm);
                rs[r] = s0[r] + s1[r];
            }
#pragma unroll
            for (int off = 1; off < 16; off <<= 1) {
#pragma unroll
                for (int r = 0; r < 8; ++r)
                    rs[r] += __shfl_xor(rs[r], off, 32);
            }
#pragma unroll
            for (int r = 0; r < 8; ++r) {
                lrun[r] = lrun[r] * corr[r] + rs[r];
                acc0[r] *= corr[r];
                acc1[r] *= corr[r];
            }

            // stage P (16q x 32k) through per-wave LDS, reload in A-layout
            __bf16* pw = p_lds[widx];
#pragma unroll
            for (int r = 0; r < 8; ++r) {
                int m = r + 8 * half;
                pw[m * 32 + n]      = (__bf16)s0[r];
                pw[m * 32 + 16 + n] = (__bf16)s1[r];
            }
            asm volatile("s_wait_dscnt 0x0" ::: "memory");
            V16 ap;
            {
                int m = lane & 15;
                ap.q[0] = *(const v4u*)(pw + m * 32 + half * 16);
                ap.q[1] = *(const v4u*)(pw + m * 32 + half * 16 + 8);
            }

            V16 bv0, bv1;
            bv0.q[0] = ((const v4u*)&vT[n][half * 16])[0];
            bv0.q[1] = ((const v4u*)&vT[n][half * 16])[1];
            bv1.q[0] = ((const v4u*)&vT[16 + n][half * 16])[0];
            bv1.q[1] = ((const v4u*)&vT[16 + n][half * 16])[1];

            acc0 = wmma_bf16(ap.b, bv0.b, acc0);
            acc1 = wmma_bf16(ap.b, bv1.b, acc1);
        }
        __syncthreads();   // tiles consumed; safe to overwrite next iteration
    }

    // epilogue: normalize, write attn output bf16 as [B,S,D] (D = h*32 + hd)
    __bf16* p = Og + ((size_t)b * SS + q0 + 8 * half) * DD + h * HD;
#pragma unroll
    for (int r = 0; r < 8; ++r) {
        float inv = 1.0f / lrun[r];
        p[(size_t)r * DD + n]      = (__bf16)(acc0[r] * inv);
        p[(size_t)r * DD + 16 + n] = (__bf16)(acc1[r] * inv);
    }
}

// ---------------------------------------------------------------------------
// Kernel 3: out = attn @ w_proj  (M=16384, K=128, N=128), f32 output.
// ---------------------------------------------------------------------------
__global__ __launch_bounds__(256) void proj_kernel(
    const __bf16* __restrict__ A, const float* __restrict__ w_proj,
    float* __restrict__ out)
{
    const int wv   = blockIdx.x * 8 + (threadIdx.x >> 5);
    const int mt   = wv >> 3;            // 1024 M-tiles
    const int nt   = wv & 7;             // 8 N-tiles
    const int m0   = mt * 16, n0 = nt * 16;
    const int lane = threadIdx.x & 31;
    const int half = lane >> 4;
    const int n    = lane & 15;

    v8f acc = {};
    const __bf16* ar = A + (size_t)(m0 + (lane & 15)) * DD;
    const float*  wc = w_proj + n0 + n;
#pragma unroll
    for (int kc = 0; kc < 4; ++kc) {
        V16 a;
        a.q[0] = *(const v4u*)(ar + kc * 32 + half * 16);
        a.q[1] = *(const v4u*)(ar + kc * 32 + half * 16 + 8);
        VF16 bf;
#pragma unroll
        for (int e = 0; e < 16; ++e)
            bf.f[e] = wc[(size_t)(kc * 32 + half * 16 + e) * DD];
        V16 bm;
        bm.b = __builtin_convertvector(bf.f, v16bf);
        acc = wmma_bf16(a.b, bm.b, acc);
    }
    float* p = out + (size_t)(m0 + 8 * half) * DD + n0 + n;
#pragma unroll
    for (int r = 0; r < 8; ++r)
        p[(size_t)r * DD] = acc[r];
}

// ---------------------------------------------------------------------------
extern "C" void kernel_launch(void* const* d_in, const int* in_sizes, int n_in,
                              void* d_out, int out_size, void* d_ws, size_t ws_size,
                              hipStream_t stream)
{
    const float* x      = (const float*)d_in[0];
    const float* w_attn = (const float*)d_in[1];
    const float* w_proj = (const float*)d_in[2];
    float*       out    = (float*)d_out;

    // workspace: 4 bf16 buffers of B*S*D = 2,097,152 elements each (16 MB total)
    const size_t NELEM = (size_t)BB * SS * DD;
    __bf16* ws = (__bf16*)d_ws;
    __bf16* Qg = ws;
    __bf16* Kg = ws + NELEM;
    __bf16* Vt = ws + 2 * NELEM;
    __bf16* Og = ws + 3 * NELEM;

    // QKV: 1024 x 24 tiles, 8 waves/block -> 3072 blocks
    qkv_kernel<<<3072, 256, 0, stream>>>(x, w_attn, Qg, Kg, Vt);
    // Attention: B*H*(S/128) blocks of 8 waves -> 512 blocks
    attn_kernel<<<512, 256, 0, stream>>>(Qg, Kg, Vt, Og);
    // Projection: 1024 x 8 tiles -> 1024 blocks
    proj_kernel<<<1024, 256, 0, stream>>>(Og, w_proj, out);
}
